// BaseRbmModel_21380347200286
// MI455X (gfx1250) — matrix operand order repaired and checked by reading, not compiled, not run on hardware
//
#include <hip/hip_runtime.h>
#include <hip/hip_bf16.h>

typedef __attribute__((ext_vector_type(16))) _Float16 v16h;
typedef __attribute__((ext_vector_type(8)))  _Float16 v8h;
typedef __attribute__((ext_vector_type(8)))  float    v8f;

#define N_VIS 88
#define N_HID 150
#define N_REC 100
#define VP 96    // padded vis (3 x 32)
#define HP 160   // padded hid (5 x 32)
#define RP 128   // padded rec (4 x 32)
#define WAVES 4
#define WSH_ELEMS (VP * HP)  // 15360 f16 per weight copy
// per-wave LDS: hbuf(2560) + vbuf(1536) + vbuf0(1536) + bhL(2560) + bvL(1536)
#define PW_ELEMS (16 * HP + 2 * (16 * VP) + 10 * 256 + 6 * 256)  // 9728 f16

// ---------- WMMA wrapper ----------
__device__ __forceinline__ v8f wmma_f16(v16h a, v16h b, v8f c) {
  return __builtin_amdgcn_wmma_f32_16x16x32_f16(false, a, false, b, (short)0, c,
                                                false, false);
}

// ---------- fragment loaders (vectorized LDS) ----------
__device__ __forceinline__ v16h loadA_lds(const _Float16* buf, int stride,
                                          int ko, int lane) {
  const int row = lane & 15;
  const int kb = (lane >> 4) << 4;
  const _Float16* p = buf + row * stride + ko + kb;
  v8h lo = *(const v8h*)(p);
  v8h hi = *(const v8h*)(p + 8);
  return __builtin_shufflevector(lo, hi, 0, 1, 2, 3, 4, 5, 6, 7, 8, 9, 10, 11,
                                 12, 13, 14, 15);
}
// v@W  : B[k=v][n=h] = WT2[n*VP + k]   (WT2 is h-major W^T, [160][96])
__device__ __forceinline__ v16h loadB_WT2(const _Float16* WT2, int t, int ko,
                                          int lane) {
  const int n = t * 16 + (lane & 15);
  const int kb2 = (lane >> 4) << 4;
  return *(const v16h*)(WT2 + n * VP + ko + kb2);
}
// h@W^T: B[k=h][n=v] = Wsh[n*HP + k]   (Wsh is v-major W, [96][160])
__device__ __forceinline__ v16h loadB_Wsh(const _Float16* Wsh, int t, int ko,
                                          int lane) {
  const int n = t * 16 + (lane & 15);
  const int kb2 = (lane >> 4) << 4;
  return *(const v16h*)(Wsh + n * HP + ko + kb2);
}
// bias GEMM B from global, branch-free: B[k=r][n] = Wy[n*N_REC + r]
__device__ __forceinline__ v16h loadB_gWy(const float* __restrict__ Wy, int t,
                                          int nMax, int ko, int lane) {
  const int n = t * 16 + (lane & 15);
  const int kb2 = (lane >> 4) << 4;
  v16h b;
#pragma unroll
  for (int j = 0; j < 16; ++j) {
    int k = ko + kb2 + j;
    bool ok = (n < nMax) && (k < N_REC);
    int idx = ok ? (n * N_REC + k) : 0;
    float v = Wy[idx];
    b[j] = (_Float16)(ok ? v : 0.f);
  }
  return b;
}

// ---------- lane-private C-fragment spill/reload in LDS (f16) ----------
__device__ __forceinline__ void storeC_f16(_Float16* base, int t, int lane, v8f c) {
  v8h h;
#pragma unroll
  for (int i = 0; i < 8; ++i) h[i] = (_Float16)c[i];
  *(v8h*)(base + (t * 32 + lane) * 8) = h;
}
__device__ __forceinline__ v8f loadC_f16(const _Float16* base, int t, int lane) {
  const v8h h = *(const v8h*)(base + (t * 32 + lane) * 8);
  v8f r;
#pragma unroll
  for (int i = 0; i < 8; ++i) r[i] = (float)h[i];
  return r;
}

// ---------- fast scalar helpers (hardware transcendentals) ----------
__device__ __forceinline__ float rng01(unsigned a, unsigned b, unsigned c) {
  unsigned x = a * 0x9E3779B9u + b * 0x85EBCA6Bu + c * 0xC2B2AE35u + 0x27220A95u;
  x ^= x >> 16; x *= 0x7FEB352Du;
  x ^= x >> 16;
  return (float)(x >> 8) * (1.0f / 16777216.0f);
}
__device__ __forceinline__ float sigm(float x) {
  return __builtin_amdgcn_rcpf(1.f + __expf(-x));  // v_exp + v_rcp
}
__device__ __forceinline__ float softplusf(float x) {
  return (x > 15.f) ? x : __logf(1.f + __expf(x));
}

__global__ void init_accum(float* a) { a[0] = 0.f; a[1] = 0.f; a[2] = 0.f; }

__global__ void finalize_kernel(const float* a, float* out) {
  float msum = a[2];
  out[0] = a[0] / msum;  // cost
  out[1] = a[1] / msum;  // monitor
}

__global__ void rbm_cd_kernel(const float* __restrict__ padded,
                              const float* __restrict__ mask,
                              const float* __restrict__ lstm,
                              const float* __restrict__ W,
                              const float* __restrict__ bv,
                              const float* __restrict__ bh,
                              const float* __restrict__ Wyv,
                              const float* __restrict__ Wyh,
                              const int* __restrict__ kptr,
                              float* __restrict__ accum, int M) {
  extern __shared__ __attribute__((aligned(32))) _Float16 smem[];
  _Float16* Wsh = smem;              // [96][160]  W   (v-major, zero-padded)
  _Float16* WT2 = smem + WSH_ELEMS;  // [160][96]  W^T (h-major, zero-padded)
  const int lane = threadIdx.x & 31;
  const int wave = threadIdx.x >> 5;
  const int half = lane >> 4;
  _Float16* wbase = smem + 2 * WSH_ELEMS + wave * PW_ELEMS;
  _Float16* hbuf = wbase;             // [16][HP] (lstm staged as [16][RP])
  _Float16* vbuf = hbuf + 16 * HP;    // [16][VP] chain state
  _Float16* vbuf0 = vbuf + 16 * VP;   // [16][VP] data (padded)
  _Float16* bhL = vbuf0 + 16 * VP;    // 10 tiles x 32 lanes x 8 (bh_t fragments)
  _Float16* bvL = bhL + 10 * 256;     // 6 tiles x 32 lanes x 8 (bv_t fragments)

  const size_t m0 = ((size_t)blockIdx.x * WAVES + wave) * 16;
  const size_t mclamp = (size_t)(M - 1);

  // ---- stage W into both layouts (block-wide, branch-free) ----
  for (int idx = threadIdx.x; idx < WSH_ELEMS; idx += blockDim.x) {
    int vv = idx / HP, hh = idx % HP;
    bool ok = (vv < N_VIS) && (hh < N_HID);
    int gi = ok ? (vv * N_HID + hh) : 0;
    float w = W[gi];
    _Float16 wh = (_Float16)(ok ? w : 0.f);
    Wsh[vv * HP + hh] = wh;
    WT2[hh * VP + vv] = wh;
  }
  // ---- stage lstm rows into hbuf as [16][128] f16 ----
  for (int idx = lane; idx < 16 * RP; idx += 32) {
    int r = idx >> 7, c = idx & 127;
    size_t m = m0 + r; m = (m > mclamp) ? mclamp : m;
    bool ok = (c < N_REC);
    size_t gi = ok ? (m * (size_t)N_REC + c) : 0;
    float v = lstm[gi];
    hbuf[idx] = (_Float16)(ok ? v : 0.f);
  }
  // ---- stage padded into vbuf (chain state) and vbuf0 (data) ----
  for (int idx = lane; idx < 16 * VP; idx += 32) {
    int r = idx / VP, c = idx % VP;
    size_t m = m0 + r; m = (m > mclamp) ? mclamp : m;
    bool ok = (c < N_VIS);
    size_t gi = ok ? (m * (size_t)N_VIS + c) : 0;
    float v = padded[gi];
    _Float16 pv = (_Float16)(ok ? v : 0.f);
    vbuf[idx] = pv;
    vbuf0[idx] = pv;
  }
  __syncthreads();

  // ---- time-dependent biases -> lane-private LDS fragments (no live VGPRs) ----
#pragma unroll
  for (int t = 0; t < 10; ++t) {
    int col = t * 16 + (lane & 15);
    float b0 = bh[(col < N_HID) ? col : 0];
    b0 = (col < N_HID) ? b0 : 0.f;
    v8f acc;
#pragma unroll
    for (int i = 0; i < 8; ++i) acc[i] = b0;
#pragma unroll
    for (int ks = 0; ks < 4; ++ks)
      acc = wmma_f16(loadA_lds(hbuf, RP, ks * 32, lane),
                     loadB_gWy(Wyh, t, N_HID, ks * 32, lane), acc);
    storeC_f16(bhL, t, lane, acc);
  }
#pragma unroll
  for (int t = 0; t < 6; ++t) {
    int col = t * 16 + (lane & 15);
    float b0 = bv[(col < N_VIS) ? col : 0];
    b0 = (col < N_VIS) ? b0 : 0.f;
    v8f acc;
#pragma unroll
    for (int i = 0; i < 8; ++i) acc[i] = b0;
#pragma unroll
    for (int ks = 0; ks < 4; ++ks)
      acc = wmma_f16(loadA_lds(hbuf, RP, ks * 32, lane),
                     loadB_gWy(Wyv, t, N_VIS, ks * 32, lane), acc);
    storeC_f16(bvL, t, lane, acc);
  }
  __syncthreads();  // hbuf is reused for h samples below

  // ---- CD-k Gibbs chain ----
  const int kk = *kptr;
  float llP[8];
#pragma unroll
  for (int i = 0; i < 8; ++i) llP[i] = 0.f;

  for (int it = 0; it < kk; ++it) {
    // h-pass: mean_h = sigmoid(v@W + bh_t); sample -> hbuf
#pragma unroll
    for (int t = 0; t < 10; ++t) {
      v8f acc = loadC_f16(bhL, t, lane);
#pragma unroll
      for (int ks = 0; ks < 3; ++ks)
        acc = wmma_f16(loadA_lds(vbuf, VP, ks * 32, lane),
                       loadB_WT2(WT2, t, ks * 32, lane), acc);
      int col = t * 16 + (lane & 15);
      float cm = (col < N_HID) ? 1.f : 0.f;
#pragma unroll
      for (int i = 0; i < 8; ++i) {
        int r = i + 8 * half;
        float mh = sigm(acc[i]);
        float u = rng01((unsigned)(m0 + r), (unsigned)col, 2u * it + 0u);
        hbuf[r * HP + col] = (_Float16)((u < mh) ? cm : 0.f);
      }
    }
    __syncthreads();
    // v-pass: mean_v = sigmoid(h@W^T + bv_t); sample -> vbuf (+ monitor last it)
    const bool lastIter = (it == kk - 1);
#pragma unroll
    for (int t = 0; t < 6; ++t) {
      v8f acc = loadC_f16(bvL, t, lane);
#pragma unroll
      for (int ks = 0; ks < 5; ++ks)
        acc = wmma_f16(loadA_lds(hbuf, HP, ks * 32, lane),
                       loadB_Wsh(Wsh, t, ks * 32, lane), acc);
      int col = t * 16 + (lane & 15);
      float cm = (col < N_VIS) ? 1.f : 0.f;
#pragma unroll
      for (int i = 0; i < 8; ++i) {
        int r = i + 8 * half;
        float mv = sigm(acc[i]);
        if (lastIter) {  // wave-uniform branch; inner math branch-free
          float p = (float)vbuf0[r * VP + col];
          llP[i] += cm * (p * __logf(mv + 1e-10f) +
                          (1.f - p) * __logf(1.f - mv + 1e-10f));
        }
        float u = rng01((unsigned)(m0 + r), (unsigned)col, 2u * it + 1u);
        vbuf[r * VP + col] = (_Float16)((u < mv) ? cm : 0.f);
      }
    }
    __syncthreads();
  }

  // ---- free energy folded into e0 = (FE(data) - FE(sample)) partials ----
  // fe(v) = -(v.bv_t) - sum softplus(v@W + bh_t)
  float e0[8];
#pragma unroll
  for (int i = 0; i < 8; ++i) e0[i] = 0.f;

#pragma unroll
  for (int t = 0; t < 10; ++t) {
    v8f c0f = loadC_f16(bhL, t, lane);
    v8f aP = c0f, aS = c0f;
#pragma unroll
    for (int ks = 0; ks < 3; ++ks) {
      v16h bb = loadB_WT2(WT2, t, ks * 32, lane);
      aP = wmma_f16(loadA_lds(vbuf0, VP, ks * 32, lane), bb, aP);
      aS = wmma_f16(loadA_lds(vbuf, VP, ks * 32, lane), bb, aS);
    }
    int col = t * 16 + (lane & 15);
    float cm = (col < N_HID) ? 1.f : 0.f;
#pragma unroll
    for (int i = 0; i < 8; ++i)
      e0[i] += cm * (softplusf(aS[i]) - softplusf(aP[i]));
  }
#pragma unroll
  for (int t = 0; t < 6; ++t) {
    v8f bvv = loadC_f16(bvL, t, lane);  // zero in padded cols by construction
    int col = t * 16 + (lane & 15);
#pragma unroll
    for (int i = 0; i < 8; ++i) {
      int r = i + 8 * half;
      e0[i] += ((float)vbuf[r * VP + col] - (float)vbuf0[r * VP + col]) * bvv[i];
    }
  }

  // ---- reduce across the 16 lanes of each half (columns) ----
#pragma unroll
  for (int off = 1; off < 16; off <<= 1) {
#pragma unroll
    for (int i = 0; i < 8; ++i) {
      e0[i] += __shfl_xor(e0[i], off, 32);
      llP[i] += __shfl_xor(llP[i], off, 32);
    }
  }

  float c0 = 0.f, c1 = 0.f, c2 = 0.f;
  if ((lane & 15) == 0) {
#pragma unroll
    for (int i = 0; i < 8; ++i) {
      size_t m = m0 + (size_t)(i + 8 * half);
      bool ok = (m < (size_t)M);
      float mk = mask[ok ? m : 0];
      mk = ok ? mk : 0.f;
      c0 += e0[i] * mk;   // (FE(padded) - FE(sample)) * mask
      c1 += llP[i] * mk;  // ll * mask
      c2 += mk;
    }
  }
  c0 += __shfl_xor(c0, 16, 32);
  c1 += __shfl_xor(c1, 16, 32);
  c2 += __shfl_xor(c2, 16, 32);
  if (lane == 0) {
    atomicAdd(&accum[0], c0);
    atomicAdd(&accum[1], c1);
    atomicAdd(&accum[2], c2);
  }
}

extern "C" void kernel_launch(void* const* d_in, const int* in_sizes, int n_in,
                              void* d_out, int out_size, void* d_ws, size_t ws_size,
                              hipStream_t stream) {
  const float* padded = (const float*)d_in[0];
  const float* mask = (const float*)d_in[1];
  const float* lstm = (const float*)d_in[2];
  const float* W = (const float*)d_in[3];
  const float* bv = (const float*)d_in[4];
  const float* bh = (const float*)d_in[5];
  const float* Wyv = (const float*)d_in[6];
  const float* Wyh = (const float*)d_in[7];
  const int* kptr = (const int*)d_in[8];
  float* out = (float*)d_out;
  float* accum = (float*)d_ws;

  const int M = in_sizes[1];  // mask has B*T elements
  const int tiles = (M + 15) / 16;
  const int blocks = (tiles + WAVES - 1) / WAVES;
  const size_t smem = (size_t)(2 * WSH_ELEMS + WAVES * PW_ELEMS) * 2;  // bytes

  init_accum<<<dim3(1), dim3(1), 0, stream>>>(accum);
  rbm_cd_kernel<<<dim3(blocks), dim3(32 * WAVES), smem, stream>>>(
      padded, mask, lstm, W, bv, bh, Wyv, Wyh, kptr, accum, M);
  finalize_kernel<<<dim3(1), dim3(1), 0, stream>>>(accum, out);
}
